// QueryGuidedMoESimple_40312563040759
// MI455X (gfx1250) — matrix operand order repaired and checked by reading, not compiled, not run on hardware
//
#include <hip/hip_runtime.h>
#include <math.h>

// ---------------------------------------------------------------------------
// QueryGuidedMoE for MI455X (gfx1250): bf16 WMMA 16x16x32, fp32 accumulate.
// B=16384, H=512, E=8, P=48 (out cols = 96), TOP_K=2.
// Compute-bound (~100 GFLOP vs ~80 MB unique traffic). All big GEMMs run on
// v_wmma_f32_16x16x32_bf16 with a fully peeled 2-stage software pipeline so
// every in-loop WMMA group waits only on its own (older) load group.
// Expert layer-1/layer-2 fused through LDS (per-wave hidden tile).
// ---------------------------------------------------------------------------

typedef __bf16 bf16_t;
typedef __bf16 v16bf __attribute__((ext_vector_type(16)));
typedef __bf16 v8bf  __attribute__((ext_vector_type(8)));
typedef float  v8f   __attribute__((ext_vector_type(8)));

union BF16x16 { v16bf v; bf16_t e[16]; unsigned short s[16]; };
union F32x8   { v8f   v; float  e[8]; };

__device__ __forceinline__ bf16_t f2bf(float f) {
    union { float f; unsigned u; } in; in.f = f;
    unsigned r = in.u + 0x7FFFu + ((in.u >> 16) & 1u);   // RNE
    union { unsigned short s; bf16_t b; } out;
    out.s = (unsigned short)(r >> 16);
    return out.b;
}
__device__ __forceinline__ float bf2f(bf16_t b) {
    union { bf16_t b; unsigned short s; } in; in.b = b;
    union { unsigned u; float f; } out;
    out.u = ((unsigned)in.s) << 16;
    return out.f;
}

// A fragment, contiguous K order for 16x32 tiles (matches B):
// lane half 0: K = k..k+15 (e0..15); half 1: K = k+16..k+31.
__device__ __forceinline__ void load_a_frag(const bf16_t* __restrict__ rowbase,
                                            int k, int hi, BF16x16& a) {
    const bf16_t* ap = rowbase + k + hi * 16;
    *(v8bf*)&a.e[0] = *(const v8bf*)ap;
    *(v8bf*)&a.e[8] = *(const v8bf*)(ap + 8);
}

// B fragments: lane = column; lanes 0-15 hold K=k..k+15, lanes 16-31 K=k+16..31
// (weights pre-transposed to [N x K] so this is one contiguous 32B load).
template <int T>
__device__ __forceinline__ void load_b_frags(const bf16_t* __restrict__ Wt, int Kd,
                                             int ncol0, int l16, int hi, int k,
                                             BF16x16* b) {
#pragma unroll
    for (int t = 0; t < T; ++t)
        b[t].v = *(const v16bf*)(Wt + (long)(ncol0 + t * 16 + l16) * Kd + k + hi * 16);
}

template <int T>
__device__ __forceinline__ void wmma_tiles(const BF16x16& a, const BF16x16* b, F32x8* acc) {
#pragma unroll
    for (int t = 0; t < T; ++t)
        acc[t].v = __builtin_amdgcn_wmma_f32_16x16x32_bf16(
            false, a.v, false, b[t].v, (short)0, acc[t].v, false, false);
}

// ---------------------------------------------------------------------------
// Weight convert + transpose: src fp32 [batch][K][N] -> dst bf16 [batch][N][K]
// ---------------------------------------------------------------------------
__global__ void moe_cvt_transpose(const float* __restrict__ src,
                                  bf16_t* __restrict__ dst,
                                  int K, int N, long total) {
    long idx = (long)blockIdx.x * blockDim.x + threadIdx.x;
    if (idx >= total) return;
    long per = (long)K * N;
    long b   = idx / per;
    long r   = idx - b * per;
    int  n   = (int)(r / K);
    int  kk  = (int)(r - (long)n * K);
    dst[idx] = f2bf(src[b * per + (long)kk * N + n]);
}

// Concat + convert activations once: Xcat[B x 1024] = [mm | q] in bf16.
__global__ void moe_cvt_concat(const float* __restrict__ mm,
                               const float* __restrict__ q,
                               bf16_t* __restrict__ Xcat, long total) {
    long idx = (long)blockIdx.x * blockDim.x + threadIdx.x;
    if (idx >= total) return;
    long i = idx >> 10;          // row
    int  c = (int)(idx & 1023);  // col in concat
    float v = (c < 512) ? mm[i * 512 + c] : q[i * 512 + (c - 512)];
    Xcat[idx] = f2bf(v);
}

// ---------------------------------------------------------------------------
// Router layer 1: Out[B x 512] = relu(Xcat @ rW1 + rb1), bf16 out.
// Block 256 thr = 8 waves; wave tile 16(M) x 64(N).
// Peeled 2-stage pipeline: loop body always prefetches k+64/k+96; drain after.
// ---------------------------------------------------------------------------
__global__ __launch_bounds__(256)
void moe_gemm_bias_relu(const bf16_t* __restrict__ A, int strideA, int K, int N,
                        const bf16_t* __restrict__ Wt,
                        const float* __restrict__ bias,
                        bf16_t* __restrict__ Out) {
    const int lane = threadIdx.x & 31;
    const int wave = threadIdx.x >> 5;
    const int hi   = lane >> 4;
    const int l16  = lane & 15;
    const int m0   = blockIdx.x * 128 + wave * 16;
    const int n0   = blockIdx.y * 64;
    const bf16_t* abase = A + (long)(m0 + l16) * strideA;

    F32x8 acc[4];
#pragma unroll
    for (int t = 0; t < 4; ++t)
#pragma unroll
        for (int j = 0; j < 8; ++j) acc[t].e[j] = 0.f;

    BF16x16 a0, a1, bA[4], bB[4];
    load_a_frag(abase, 0, hi, a0);
    load_b_frags<4>(Wt, K, n0, l16, hi, 0, bA);
    load_a_frag(abase, 32, hi, a1);
    load_b_frags<4>(Wt, K, n0, l16, hi, 32, bB);
    int k = 0;
    for (; k < K - 64; k += 64) {                // K % 64 == 0, K >= 128
        wmma_tiles<4>(a0, bA, acc);              // waits on own group only
        load_a_frag(abase, k + 64, hi, a0);
        load_b_frags<4>(Wt, K, n0, l16, hi, k + 64, bA);
        __builtin_prefetch((const void*)(Wt + (long)(n0 + l16) * K + k + 192), 0, 1);
        wmma_tiles<4>(a1, bB, acc);
        load_a_frag(abase, k + 96, hi, a1);
        load_b_frags<4>(Wt, K, n0, l16, hi, k + 96, bB);
    }
    wmma_tiles<4>(a0, bA, acc);                  // drain
    wmma_tiles<4>(a1, bB, acc);

#pragma unroll
    for (int t = 0; t < 4; ++t) {
        const int col = n0 + t * 16 + l16;
        const float bv = bias[col];
#pragma unroll
        for (int r = 0; r < 8; ++r) {
            const int row = m0 + hi * 8 + r;
            float v = acc[t].e[r] + bv;
            v = v > 0.f ? v : 0.f;
            Out[(long)row * N + col] = f2bf(v);
        }
    }
}

// ---------------------------------------------------------------------------
// Fused expert: h = relu(X @ W1 + b1) staged per-wave in LDS (bf16), then
// Out[B x 96] (+)= gate[:,e] * (h @ W2 + b2).
// Block 128 thr = 4 waves = 64 rows; LDS 64x512 bf16 = 64KB.
// ---------------------------------------------------------------------------
__global__ __launch_bounds__(128)
void moe_expert_fused(const bf16_t* __restrict__ Xcat,  // [B x 1024], cols 0..511 = mm
                      const bf16_t* __restrict__ W1t,   // [512 x 512]
                      const float*  __restrict__ b1,    // [512]
                      const bf16_t* __restrict__ W2t,   // [96 x 512]
                      const float*  __restrict__ b2,    // [96]
                      const float*  __restrict__ gate,  // [B x 8]
                      int e, int accumulate,
                      float* __restrict__ OutAcc) {     // [B x 96]
    constexpr int K = 512, N2 = 96, SA = 1024;
    __shared__ bf16_t hsh[64 * 512];                    // 64KB

    const int lane = threadIdx.x & 31;
    const int wave = threadIdx.x >> 5;
    const int hi   = lane >> 4;
    const int l16  = lane & 15;
    const int m0   = blockIdx.x * 64 + wave * 16;       // global row base of wave
    const int lr   = wave * 16;                         // LDS row base of wave
    const bf16_t* abase = Xcat + (long)(m0 + l16) * SA; // first 512 cols = mm

    // ---- Stage 1: hidden tile 16 x 512 -> LDS ----
    for (int nc = 0; nc < 512; nc += 64) {
        F32x8 acc[4];
#pragma unroll
        for (int t = 0; t < 4; ++t)
#pragma unroll
            for (int j = 0; j < 8; ++j) acc[t].e[j] = 0.f;

        BF16x16 a0, a1, bA[4], bB[4];
        load_a_frag(abase, 0, hi, a0);
        load_b_frags<4>(W1t, K, nc, l16, hi, 0, bA);
        load_a_frag(abase, 32, hi, a1);
        load_b_frags<4>(W1t, K, nc, l16, hi, 32, bB);
        int k = 0;
        for (; k < K - 64; k += 64) {
            wmma_tiles<4>(a0, bA, acc);
            load_a_frag(abase, k + 64, hi, a0);
            load_b_frags<4>(W1t, K, nc, l16, hi, k + 64, bA);
            __builtin_prefetch((const void*)(W1t + (long)(nc + l16) * K + k + 192), 0, 1);
            wmma_tiles<4>(a1, bB, acc);
            load_a_frag(abase, k + 96, hi, a1);
            load_b_frags<4>(W1t, K, nc, l16, hi, k + 96, bB);
        }
        wmma_tiles<4>(a0, bA, acc);
        wmma_tiles<4>(a1, bB, acc);

#pragma unroll
        for (int t = 0; t < 4; ++t) {
            const int col = nc + t * 16 + l16;
            const float bv = b1[col];
#pragma unroll
            for (int r = 0; r < 8; ++r) {
                float v = acc[t].e[r] + bv;
                v = v > 0.f ? v : 0.f;
                hsh[(lr + hi * 8 + r) * 512 + col] = f2bf(v);
            }
        }
    }

    __syncthreads();   // tiles are wave-private, but order DS stores vs. loads

    // ---- Stage 2: out tile 16 x 96 from LDS hidden (pipelined as well) ----
    F32x8 acc2[6];
#pragma unroll
    for (int t = 0; t < 6; ++t)
#pragma unroll
        for (int j = 0; j < 8; ++j) acc2[t].e[j] = 0.f;

    const bf16_t* lbase = &hsh[(lr + l16) * 512];
    BF16x16 a0, a1, bA[6], bB[6];
    load_a_frag(lbase, 0, hi, a0);                      // ds_load_b128 x2
    load_b_frags<6>(W2t, K, 0, l16, hi, 0, bA);
    load_a_frag(lbase, 32, hi, a1);
    load_b_frags<6>(W2t, K, 0, l16, hi, 32, bB);
    int k = 0;
    for (; k < K - 64; k += 64) {
        wmma_tiles<6>(a0, bA, acc2);
        load_a_frag(lbase, k + 64, hi, a0);
        load_b_frags<6>(W2t, K, 0, l16, hi, k + 64, bA);
        wmma_tiles<6>(a1, bB, acc2);
        load_a_frag(lbase, k + 96, hi, a1);
        load_b_frags<6>(W2t, K, 0, l16, hi, k + 96, bB);
    }
    wmma_tiles<6>(a0, bA, acc2);
    wmma_tiles<6>(a1, bB, acc2);

    float g[8];
#pragma unroll
    for (int r = 0; r < 8; ++r)
        g[r] = gate[(long)(m0 + hi * 8 + r) * 8 + e];

#pragma unroll
    for (int t = 0; t < 6; ++t) {
        const int col = t * 16 + l16;
        const float bv = b2[col];
#pragma unroll
        for (int r = 0; r < 8; ++r) {
            const long o = (long)(m0 + hi * 8 + r) * N2 + col;
            const float v = (acc2[t].e[r] + bv) * g[r];
            OutAcc[o] = accumulate ? (OutAcc[o] + v) : v;
        }
    }
}

// ---------------------------------------------------------------------------
// Router head: logits = Hr @ rW2 + rb2, softmax over 8, top-2 renorm -> gate.
// ---------------------------------------------------------------------------
__global__ void moe_router_gate(const bf16_t* __restrict__ Hr,
                                const float* __restrict__ rW2,
                                const float* __restrict__ rb2,
                                float* __restrict__ gate, int B) {
    const int i = blockIdx.x * blockDim.x + threadIdx.x;
    if (i >= B) return;
    float acc[8];
#pragma unroll
    for (int e = 0; e < 8; ++e) acc[e] = rb2[e];
    for (int k = 0; k < 512; ++k) {
        const float hv = bf2f(Hr[(long)i * 512 + k]);
        const float4 w0 = *(const float4*)(rW2 + (long)k * 8);
        const float4 w1 = *(const float4*)(rW2 + (long)k * 8 + 4);
        acc[0] += hv * w0.x; acc[1] += hv * w0.y; acc[2] += hv * w0.z; acc[3] += hv * w0.w;
        acc[4] += hv * w1.x; acc[5] += hv * w1.y; acc[6] += hv * w1.z; acc[7] += hv * w1.w;
    }
    float mx = acc[0];
#pragma unroll
    for (int e = 1; e < 8; ++e) mx = acc[e] > mx ? acc[e] : mx;
    float sum = 0.f, w[8];
#pragma unroll
    for (int e = 0; e < 8; ++e) { w[e] = expf(acc[e] - mx); sum += w[e]; }
    const float inv = 1.f / sum;
#pragma unroll
    for (int e = 0; e < 8; ++e) w[e] *= inv;
    int i1 = 0;
#pragma unroll
    for (int e = 1; e < 8; ++e) if (w[e] > w[i1]) i1 = e;
    int i2 = (i1 == 0) ? 1 : 0;
#pragma unroll
    for (int e = 0; e < 8; ++e) if (e != i1 && w[e] > w[i2]) i2 = e;
    const float s = 1.f / (w[i1] + w[i2] + 1e-6f);
#pragma unroll
    for (int e = 0; e < 8; ++e) gate[(long)i * 8 + e] = 0.f;
    gate[(long)i * 8 + i1] = w[i1] * s;
    gate[(long)i * 8 + i2] += w[i2] * s;
}

__global__ void moe_sigmoid_inplace(float* __restrict__ p, long n) {
    long idx = (long)blockIdx.x * blockDim.x + threadIdx.x;
    if (idx >= n) return;
    p[idx] = 1.f / (1.f + expf(-p[idx]));
}

// ---------------------------------------------------------------------------
// Launch
// ---------------------------------------------------------------------------
extern "C" void kernel_launch(void* const* d_in, const int* in_sizes, int n_in,
                              void* d_out, int out_size, void* d_ws, size_t ws_size,
                              hipStream_t stream) {
    const int H = 512, E = 8, P2 = 96;
    const long B = (long)in_sizes[0] / H;   // 16384

    const float* mm  = (const float*)d_in[0];
    const float* qf  = (const float*)d_in[1];
    const float* rW1 = (const float*)d_in[2];
    const float* rb1 = (const float*)d_in[3];
    const float* rW2 = (const float*)d_in[4];
    const float* rb2 = (const float*)d_in[5];
    const float* eW1 = (const float*)d_in[6];
    const float* eb1 = (const float*)d_in[7];
    const float* eW2 = (const float*)d_in[8];
    const float* eb2 = (const float*)d_in[9];
    float* out = (float*)d_out;

    // Workspace carve-up (256B aligned).
    char*  w   = (char*)d_ws;
    size_t off = 0;
    auto alloc = [&](size_t bytes) { char* p = w + off; off += (bytes + 255) & ~(size_t)255; return p; };
    bf16_t* rW1t = (bf16_t*)alloc((size_t)H * (2 * H) * 2);   // [512 x 1024]
    bf16_t* eW1t = (bf16_t*)alloc((size_t)E * H * H * 2);     // [8][512 x 512]
    bf16_t* eW2t = (bf16_t*)alloc((size_t)E * P2 * H * 2);    // [8][96 x 512]
    bf16_t* Xcat = (bf16_t*)alloc((size_t)B * (2 * H) * 2);   // [B x 1024]
    bf16_t* Hr   = (bf16_t*)alloc((size_t)B * H * 2);         // router hidden
    float*  gate = (float*) alloc((size_t)B * E * 4);         // [B x 8]
    (void)ws_size; (void)n_in; (void)out_size;

    // 1) Weights -> bf16 transposed [N x K]; activations -> bf16 concat (once).
    {
        long t0 = (long)H * (2 * H);
        moe_cvt_transpose<<<dim3((t0 + 255) / 256), dim3(256), 0, stream>>>(rW1, rW1t, 2 * H, H, t0);
        long t1 = (long)E * H * H;
        moe_cvt_transpose<<<dim3((t1 + 255) / 256), dim3(256), 0, stream>>>(eW1, eW1t, H, H, t1);
        long t2 = (long)E * P2 * H;
        moe_cvt_transpose<<<dim3((t2 + 255) / 256), dim3(256), 0, stream>>>(eW2, eW2t, H, P2, t2);
        long t3 = B * (2 * H);
        moe_cvt_concat<<<dim3((t3 + 255) / 256), dim3(256), 0, stream>>>(mm, qf, Xcat, t3);
    }

    // 2) Router layer 1 -> Hr (bf16).
    moe_gemm_bias_relu<<<dim3(B / 128, H / 64), dim3(256), 0, stream>>>(
        Xcat, /*strideA=*/2 * H, /*K=*/2 * H, /*N=*/H, rW1t, rb1, Hr);

    // 3) Router layer 2 + softmax + top-2 -> gate.
    moe_router_gate<<<dim3((B + 255) / 256), dim3(256), 0, stream>>>(Hr, rW2, rb2, gate, (int)B);

    // 4) Experts: fused L1(ReLU)->LDS->L2, gate-weighted accumulate into out.
    for (int e = 0; e < E; ++e) {
        moe_expert_fused<<<dim3(B / 64), dim3(128), 0, stream>>>(
            Xcat, eW1t + (size_t)e * H * H, eb1 + (size_t)e * H,
            eW2t + (size_t)e * P2 * H, eb2 + (size_t)e * P2,
            gate, e, (e > 0) ? 1 : 0, out);
    }

    // 5) Sigmoid in place.
    {
        long n = B * P2;
        moe_sigmoid_inplace<<<dim3((n + 255) / 256), dim3(256), 0, stream>>>(out, n);
    }
}
